// AML_GraphNetwork_69595650064479
// MI455X (gfx1250) — compile-verified, matrix operand order checked by reading, not run on hardware
//
#include <hip/hip_runtime.h>
#include <hip/hip_bf16.h>
#include <math.h>

#define NN   50000
#define EE   800000
#define INC  128
#define HIDC 32
#define HEADS 4
#define OUTC 2
#define ETOT (EE + NN)          // edges + self loops
#define NEGS 0.2f
#define BNEPS 1e-5f

typedef __attribute__((ext_vector_type(16))) _Float16 v16h;
typedef __attribute__((ext_vector_type(8)))  _Float16 v8h;
typedef __attribute__((ext_vector_type(8)))  float    v8f;

// ---------- ordered-int encoding for float atomicMax ----------
__device__ __forceinline__ int f2o(float f) {
    int i = __float_as_int(f);
    return (i >= 0) ? i : (i ^ 0x7FFFFFFF);
}
__device__ __forceinline__ float o2f(int i) {
    return (i >= 0) ? __int_as_float(i) : __int_as_float(i ^ 0x7FFFFFFF);
}
#define NEG_INF_ORD 0x807FFFFF   // f2o(-inf)

// ---------- init helpers ----------
__global__ void k_zero_f(float* p, int n) {
    int t = blockIdx.x * blockDim.x + threadIdx.x;
    if (t < n) p[t] = 0.0f;
}
__global__ void k_fill_i(int* p, int n, int v) {
    int t = blockIdx.x * blockDim.x + threadIdx.x;
    if (t < n) p[t] = v;
}

// ---------- mean of edge_attr (E x 1) ----------
__global__ void k_sum_attr(const float* __restrict__ ea, int n, float* out) {
    __shared__ float sm[256];
    float s = 0.0f;
    for (int i = blockIdx.x * blockDim.x + threadIdx.x; i < n;
         i += gridDim.x * blockDim.x)
        s += ea[i];
    sm[threadIdx.x] = s;
    __syncthreads();
    for (int o = 128; o > 0; o >>= 1) {
        if (threadIdx.x < o) sm[threadIdx.x] += sm[threadIdx.x + o];
        __syncthreads();
    }
    if (threadIdx.x == 0) atomicAdd(out, sm[0]);
}

// ---------- WMMA GEMM: Y[M,NOUT] = X[M,K] @ W[K,NOUT] + bias ----------
// Weights staged to LDS as f16, transposed to [NOUT][K] so each lane's B
// fragment is two contiguous 16B ds reads. One wave32 per 16-row strip of X;
// A fragments (all of K) loaded once via float4 and reused across all N-tiles.
template <int K, int NOUT>
__global__ void k_gemm_wmma(const float* __restrict__ X, const float* __restrict__ W,
                            const float* __restrict__ bias, float* __restrict__ Y,
                            int nstrips) {
    __shared__ _Float16 Wt[NOUT * K];
    for (int i = threadIdx.x; i < K * NOUT; i += blockDim.x) {
        int k = i / NOUT, n = i - k * NOUT;         // coalesced global read
        Wt[n * K + k] = (_Float16)W[i];             // transposed LDS write
    }
    __syncthreads();

    int strip = blockIdx.x * 4 + (threadIdx.x >> 5);
    if (strip >= nstrips) return;                   // wave-uniform guard
    int lane = threadIdx.x & 31;
    int half = lane >> 4;                           // 0: K 0-7/16-23, 1: K 8-15/24-31
    int idx  = lane & 15;                           // A row / B col within tile
    int m0 = strip << 4;

    // load all A fragments for this 16-row strip (float4 -> b128 loads)
    v16h A[K / 32];
    const float* xrow = X + (long)(m0 + idx) * K;
#pragma unroll
    for (int kf = 0; kf < K / 32; ++kf) {
        int kb = kf * 32 + half * 8;
        float4 l0 = *(const float4*)(xrow + kb);
        float4 l1 = *(const float4*)(xrow + kb + 4);
        float4 h0 = *(const float4*)(xrow + kb + 16);
        float4 h1 = *(const float4*)(xrow + kb + 20);
        A[kf][0]  = (_Float16)l0.x; A[kf][1]  = (_Float16)l0.y;
        A[kf][2]  = (_Float16)l0.z; A[kf][3]  = (_Float16)l0.w;
        A[kf][4]  = (_Float16)l1.x; A[kf][5]  = (_Float16)l1.y;
        A[kf][6]  = (_Float16)l1.z; A[kf][7]  = (_Float16)l1.w;
        A[kf][8]  = (_Float16)h0.x; A[kf][9]  = (_Float16)h0.y;
        A[kf][10] = (_Float16)h0.z; A[kf][11] = (_Float16)h0.w;
        A[kf][12] = (_Float16)h1.x; A[kf][13] = (_Float16)h1.y;
        A[kf][14] = (_Float16)h1.z; A[kf][15] = (_Float16)h1.w;
    }

#pragma unroll
    for (int nt = 0; nt < NOUT / 16; ++nt) {
        int n0 = nt << 4;
        const _Float16* wcol = &Wt[(n0 + idx) * K];
        v8f c = {};
#pragma unroll
        for (int kf = 0; kf < K / 32; ++kf) {
            int kb = kf * 32 + half * 8;
            v8h lo = *(const v8h*)(wcol + kb);       // 16B ds_load
            v8h hi = *(const v8h*)(wcol + kb + 16);  // 16B ds_load
            v16h b;
#pragma unroll
            for (int i = 0; i < 8; ++i) { b[i] = lo[i]; b[8 + i] = hi[i]; }
            c = __builtin_amdgcn_wmma_f32_16x16x32_f16(false, A[kf], false, b,
                                                       (short)0, c, false, false);
        }
        float bv = bias[n0 + idx];
#pragma unroll
        for (int v = 0; v < 8; ++v) {
            int row = v + 8 * half;                  // D layout: lanes 0-15 M=0..7, 16-31 M=8..15
            Y[(long)(m0 + row) * NOUT + n0 + idx] = c[v] + bv;
        }
    }
}

// ---------- edge pass 1: logits + segment max (float4 gathers) ----------
__global__ void k_edge_logits(const int* __restrict__ ei, const float* __restrict__ eattr,
                              const float* __restrict__ attrSum,
                              const float* __restrict__ xl, const float* __restrict__ xr,
                              const float* __restrict__ We, const float* __restrict__ att,
                              float* __restrict__ logits, int* __restrict__ maxb,
                              int H, int C) {
    int t = blockIdx.x * blockDim.x + threadIdx.x;
    if (t >= ETOT * H) return;
    int e = t / H, h = t - e * H;
    int s, d; float ea;
    if (e < EE) { s = ei[e]; d = ei[EE + e]; ea = eattr[e]; }
    else        { s = d = e - EE; ea = attrSum[0] * (1.0f / EE); }
    const float4* xls = (const float4*)(xl + (long)s * H * C + h * C);
    const float4* xrd = (const float4*)(xr + (long)d * H * C + h * C);
    const float4* we  = (const float4*)(We + h * C);
    const float4* at  = (const float4*)(att + h * C);
    float acc = 0.0f;
    for (int q = 0; q < C / 4; ++q) {
        float4 a4 = xls[q], b4 = xrd[q], w4 = we[q], t4 = at[q];
        float v0 = a4.x + b4.x + ea * w4.x;
        float v1 = a4.y + b4.y + ea * w4.y;
        float v2 = a4.z + b4.z + ea * w4.z;
        float v3 = a4.w + b4.w + ea * w4.w;
        v0 = (v0 > 0.0f) ? v0 : NEGS * v0;
        v1 = (v1 > 0.0f) ? v1 : NEGS * v1;
        v2 = (v2 > 0.0f) ? v2 : NEGS * v2;
        v3 = (v3 > 0.0f) ? v3 : NEGS * v3;
        acc += v0 * t4.x + v1 * t4.y + v2 * t4.z + v3 * t4.w;
    }
    logits[t] = acc;
    atomicMax(maxb + d * H + h, f2o(acc));
}

// ---------- edge pass 2: exp + segment sum ----------
__global__ void k_edge_exp(const int* __restrict__ ei, float* __restrict__ logits,
                           const int* __restrict__ maxb, float* __restrict__ denom, int H) {
    int t = blockIdx.x * blockDim.x + threadIdx.x;
    if (t >= ETOT * H) return;
    int e = t / H, h = t - e * H;
    int d = (e < EE) ? ei[EE + e] : (e - EE);
    float a = __expf(logits[t] - o2f(maxb[d * H + h]));
    logits[t] = a;
    atomicAdd(denom + d * H + h, a);
}

// ---------- edge pass 3: alpha-weighted scatter-add ----------
__global__ void k_edge_agg(const int* __restrict__ ei, const float* __restrict__ a,
                           const float* __restrict__ denom, const float* __restrict__ xl,
                           float* __restrict__ outAcc, int H, int C) {
    int t = blockIdx.x * blockDim.x + threadIdx.x;
    if (t >= ETOT * H) return;
    int e = t / H, h = t - e * H;
    int s, d;
    if (e < EE) { s = ei[e]; d = ei[EE + e]; }
    else        { s = d = e - EE; }
    float alpha = a[t] / denom[d * H + h];
    const float4* sp = (const float4*)(xl + (long)s * H * C + h * C);
    float* dp = outAcc + (long)d * H * C + h * C;
    for (int q = 0; q < C / 4; ++q) {
        float4 v4 = sp[q];
        atomicAdd(dp + 4 * q + 0, alpha * v4.x);
        atomicAdd(dp + 4 * q + 1, alpha * v4.y);
        atomicAdd(dp + 4 * q + 2, alpha * v4.z);
        atomicAdd(dp + 4 * q + 3, alpha * v4.w);
    }
}

// ---------- fused bias + batchnorm(eval) + relu, in place ----------
__global__ void k_bias_bn_relu(float* __restrict__ v, const float* __restrict__ bias,
                               const float* __restrict__ gamma, const float* __restrict__ beta,
                               const float* __restrict__ mean, const float* __restrict__ var,
                               int n, int F) {
    int t = blockIdx.x * blockDim.x + threadIdx.x;
    if (t >= n) return;
    int f = t % F;
    float x = v[t] + bias[f];
    x = (x - mean[f]) * (gamma[f] * rsqrtf(var[f] + BNEPS)) + beta[f];
    v[t] = fmaxf(x, 0.0f);
}

// ---------- classifier: out[N,2] = h[N,32] @ W[32,2] + b ----------
__global__ void k_classifier(const float* __restrict__ h, const float* __restrict__ W,
                             const float* __restrict__ b, float* __restrict__ out) {
    int t = blockIdx.x * blockDim.x + threadIdx.x;
    if (t >= NN * OUTC) return;
    int n = t / OUTC, o = t - n * OUTC;
    float acc = b[o];
    const float* hp = h + n * HIDC;
    for (int c = 0; c < HIDC; ++c) acc += hp[c] * W[c * OUTC + o];
    out[t] = acc;
}

static inline int cdiv(long a, int b) { return (int)((a + b - 1) / b); }

extern "C" void kernel_launch(void* const* d_in, const int* in_sizes, int n_in,
                              void* d_out, int out_size, void* d_ws, size_t ws_size,
                              hipStream_t stream) {
    const float* x      = (const float*)d_in[0];
    const int*   ei     = (const int*)  d_in[1];
    const float* eattr  = (const float*)d_in[2];
    const float* c1_Wl  = (const float*)d_in[3];
    const float* c1_bl  = (const float*)d_in[4];
    const float* c1_Wr  = (const float*)d_in[5];
    const float* c1_br  = (const float*)d_in[6];
    const float* c1_We  = (const float*)d_in[7];
    const float* c1_att = (const float*)d_in[8];
    const float* c1_bias= (const float*)d_in[9];
    const float* bn1_g  = (const float*)d_in[10];
    const float* bn1_b  = (const float*)d_in[11];
    const float* bn1_m  = (const float*)d_in[12];
    const float* bn1_v  = (const float*)d_in[13];
    const float* c2_Wl  = (const float*)d_in[14];
    const float* c2_bl  = (const float*)d_in[15];
    const float* c2_Wr  = (const float*)d_in[16];
    const float* c2_br  = (const float*)d_in[17];
    const float* c2_We  = (const float*)d_in[18];
    const float* c2_att = (const float*)d_in[19];
    const float* c2_bias= (const float*)d_in[20];
    const float* bn2_g  = (const float*)d_in[21];
    const float* bn2_b  = (const float*)d_in[22];
    const float* bn2_m  = (const float*)d_in[23];
    const float* bn2_v  = (const float*)d_in[24];
    const float* clf_W  = (const float*)d_in[25];
    const float* clf_b  = (const float*)d_in[26];
    float* out = (float*)d_out;

    // workspace layout (floats)
    float* ws = (float*)d_ws;
    long o = 0;
    float* xl1    = ws + o; o += (long)NN * 128;
    float* xr1    = ws + o; o += (long)NN * 128;
    float* agg1   = ws + o; o += (long)NN * 128;
    float* xl2    = ws + o; o += (long)NN * 32;
    float* xr2    = ws + o; o += (long)NN * 32;
    float* agg2   = ws + o; o += (long)NN * 32;
    float* logits = ws + o; o += (long)ETOT * HEADS;
    int*   maxb   = (int*)(ws + o); o += (long)NN * HEADS;
    float* denom  = ws + o; o += (long)NN * HEADS;
    float* attrSum= ws + o; o += 1;

    const int TB = 256;
    const int NSTRIPS = NN / 16;   // 3125, exact

    // ---- edge_attr mean ----
    k_zero_f<<<1, 32, 0, stream>>>(attrSum, 1);
    k_sum_attr<<<256, TB, 0, stream>>>(eattr, EE, attrSum);

    // ---- layer 1 dense transforms (WMMA, LDS-staged weights) ----
    k_gemm_wmma<128, 128><<<cdiv(NSTRIPS, 4), 128, 0, stream>>>(x, c1_Wl, c1_bl, xl1, NSTRIPS);
    k_gemm_wmma<128, 128><<<cdiv(NSTRIPS, 4), 128, 0, stream>>>(x, c1_Wr, c1_br, xr1, NSTRIPS);

    // ---- layer 1 message passing ----
    k_fill_i<<<cdiv((long)NN * HEADS, TB), TB, 0, stream>>>(maxb, NN * HEADS, NEG_INF_ORD);
    k_zero_f<<<cdiv((long)NN * HEADS, TB), TB, 0, stream>>>(denom, NN * HEADS);
    k_zero_f<<<cdiv((long)NN * 128, TB), TB, 0, stream>>>(agg1, NN * 128);

    int et1 = ETOT * HEADS;
    k_edge_logits<<<cdiv(et1, TB), TB, 0, stream>>>(ei, eattr, attrSum, xl1, xr1,
                                                    c1_We, c1_att, logits, maxb, HEADS, HIDC);
    k_edge_exp<<<cdiv(et1, TB), TB, 0, stream>>>(ei, logits, maxb, denom, HEADS);
    k_edge_agg<<<cdiv(et1, TB), TB, 0, stream>>>(ei, logits, denom, xl1, agg1, HEADS, HIDC);
    k_bias_bn_relu<<<cdiv((long)NN * 128, TB), TB, 0, stream>>>(agg1, c1_bias, bn1_g, bn1_b,
                                                                bn1_m, bn1_v, NN * 128, 128);

    // ---- layer 2 dense transforms (WMMA) ----
    k_gemm_wmma<128, 32><<<cdiv(NSTRIPS, 4), 128, 0, stream>>>(agg1, c2_Wl, c2_bl, xl2, NSTRIPS);
    k_gemm_wmma<128, 32><<<cdiv(NSTRIPS, 4), 128, 0, stream>>>(agg1, c2_Wr, c2_br, xr2, NSTRIPS);

    // ---- layer 2 message passing (H=1, C=32) ----
    k_fill_i<<<cdiv((long)NN, TB), TB, 0, stream>>>(maxb, NN, NEG_INF_ORD);
    k_zero_f<<<cdiv((long)NN, TB), TB, 0, stream>>>(denom, NN);
    k_zero_f<<<cdiv((long)NN * 32, TB), TB, 0, stream>>>(agg2, NN * 32);

    int et2 = ETOT;
    k_edge_logits<<<cdiv(et2, TB), TB, 0, stream>>>(ei, eattr, attrSum, xl2, xr2,
                                                    c2_We, c2_att, logits, maxb, 1, HIDC);
    k_edge_exp<<<cdiv(et2, TB), TB, 0, stream>>>(ei, logits, maxb, denom, 1);
    k_edge_agg<<<cdiv(et2, TB), TB, 0, stream>>>(ei, logits, denom, xl2, agg2, 1, HIDC);
    k_bias_bn_relu<<<cdiv((long)NN * 32, TB), TB, 0, stream>>>(agg2, c2_bias, bn2_g, bn2_b,
                                                               bn2_m, bn2_v, NN * 32, 32);

    // ---- classifier ----
    k_classifier<<<cdiv((long)NN * OUTC, TB), TB, 0, stream>>>(agg2, clf_W, clf_b, out);
    (void)in_sizes; (void)n_in; (void)out_size; (void)ws_size;
}
